// Autoformer_76708115906663
// MI455X (gfx1250) — matrix operand order repaired and checked by reading, not compiled, not run on hardware
//
#include <hip/hip_runtime.h>
#include <hip/hip_bf16.h>
#include <math.h>

typedef __attribute__((ext_vector_type(16))) _Float16 v16h;
typedef __attribute__((ext_vector_type(8)))  float    v8f;

// ---------------------------------------------------------------------------
// WMMA fragment helpers (CDNA5 16x16x32 f16 layout, wave32)
// A (16x32): lane&15 = row M; lane[4] selects K 0..15 / 16..31 (contiguous)
// B (32x16): lane&15 = col N; lane[4] selects K 0..15 / 16..31 (contiguous)
// C/D v8f:   reg r -> M = r + 8*lane[4], N = lane&15
// ---------------------------------------------------------------------------
__device__ __forceinline__ v16h af_load_a(const _Float16* __restrict__ arow, int k0, int hi) {
    v16h a;
    const int kaLo = hi ? 16 : 0;
    const int kaHi = kaLo + 8;
#pragma unroll
    for (int j = 0; j < 4; ++j) {
        a[2 * j]     = arow[k0 + kaLo + 2 * j];
        a[2 * j + 1] = arow[k0 + kaLo + 2 * j + 1];
        a[8 + 2 * j]     = arow[k0 + kaHi + 2 * j];
        a[8 + 2 * j + 1] = arow[k0 + kaHi + 2 * j + 1];
    }
    return a;
}

#define AF_WMMA(Afrag, Bfrag, Cacc) \
    __builtin_amdgcn_wmma_f32_16x16x32_f16(false, (Afrag), false, (Bfrag), (short)0, (Cacc), false, false)

// ---------------------------------------------------------------------------
// Generic GEMM:  C[M,N] = act(A[M,K] * W[N,K]^T + bias) (+ resid)
// K is a template constant (256 or 1024) so the software-pipelined k-loop
// fully unrolls: double-buffer rotation becomes SSA renaming (no v_mov
// chains / branches), loads for chunk k+1 issue ahead of chunk k's WMMAs.
// Workgroup: 8 waves -> 64(M) x 64(N); each wave owns a 16x32 tile
// (two WMMAs sharing one A fragment).
// M%64==0, N%64==0 guaranteed by caller.
// ---------------------------------------------------------------------------
template <int K>
__global__ __launch_bounds__(256) void k_gemm_wmma(
    const _Float16* __restrict__ A, const _Float16* __restrict__ W,
    const float* __restrict__ bias, const float* __restrict__ resid,
    float* __restrict__ outF, _Float16* __restrict__ outH,
    int M, int N, int applyGelu)
{
    const int wave = threadIdx.x >> 5;
    const int lane = threadIdx.x & 31;
    const int wm = wave & 3;          // 0..3  -> M sub-tile
    const int wn = wave >> 2;         // 0..1  -> 32-wide N sub-tile
    const int tilesN = N >> 6;
    const int tM = blockIdx.x / tilesN;
    const int tN = blockIdx.x % tilesN;
    const int row0 = tM * 64 + wm * 16;
    const int col0 = tN * 64 + wn * 32;
    const int r  = lane & 15;
    const int hi = lane >> 4;

    const _Float16* __restrict__ arow  = A + (size_t)(row0 + r) * K;
    const _Float16* __restrict__ brow0 = W + (size_t)(col0 + r) * K + hi * 16;
    const _Float16* __restrict__ brow1 = W + (size_t)(col0 + 16 + r) * K + hi * 16;

    v8f acc0 = {}, acc1 = {};
    // prologue: fragments for k0 = 0
    v16h a  = af_load_a(arow, 0, hi);
    v16h b0 = *(const v16h*)(brow0);
    v16h b1 = *(const v16h*)(brow1);

#pragma unroll
    for (int k0 = 0; k0 < K - 32; k0 += 32) {
        const int kn = k0 + 32;
        v16h an  = af_load_a(arow, kn, hi);        // next chunk in flight
        v16h b0n = *(const v16h*)(brow0 + kn);
        v16h b1n = *(const v16h*)(brow1 + kn);
        if (kn + 32 < K) {                         // deep-K: stream weights ahead
            __builtin_prefetch((const void*)(brow0 + kn + 32), 0, 1);
            __builtin_prefetch((const void*)(brow1 + kn + 32), 0, 1);
        }
        acc0 = AF_WMMA(a, b0, acc0);
        acc1 = AF_WMMA(a, b1, acc1);
        a = an; b0 = b0n; b1 = b1n;
    }
    // epilogue k-step
    acc0 = AF_WMMA(a, b0, acc0);
    acc1 = AF_WMMA(a, b1, acc1);

    const int colA = col0 + r;
    const int colB = col0 + 16 + r;
    const float bvA = bias ? bias[colA] : 0.0f;
    const float bvB = bias ? bias[colB] : 0.0f;
#pragma unroll
    for (int t = 0; t < 8; ++t) {
        const int rowm = row0 + t + (hi ? 8 : 0);
        float v0 = acc0[t] + bvA;
        float v1 = acc1[t] + bvB;
        if (applyGelu) {
            v0 = 0.5f * v0 * (1.0f + erff(v0 * 0.70710678118f));
            v1 = 0.5f * v1 * (1.0f + erff(v1 * 0.70710678118f));
        }
        const size_t o0 = (size_t)rowm * N + colA;
        const size_t o1 = (size_t)rowm * N + colB;
        if (resid) { v0 += resid[o0]; v1 += resid[o1]; }
        if (outF) { outF[o0] = v0; outF[o1] = v1; }
        if (outH) { outH[o0] = (_Float16)v0; outH[o1] = (_Float16)v1; }
    }
}

// ---------------------------------------------------------------------------
// Channel-mean circular correlation via Gram-matrix diagonal sums.
// meanv[b,tau] += sum_t <Q[b,t,:256], K[b,(t-tau)%L,:256]>
// One workgroup = one 16-row t-block. The 8 Q fragments (full K=256) are
// hoisted into registers once; waves stripe over s-tile *pairs* so each
// inner step is 2 B-loads + 2 WMMAs. Per-lag partials reduced in LDS
// (ds_add_f32), then one global atomic pass.
// ---------------------------------------------------------------------------
__global__ __launch_bounds__(256) void k_corr_wmma(
    const _Float16* __restrict__ Q, const _Float16* __restrict__ Km,
    float* __restrict__ meanv, int L, int strideQ, int strideK)
{
    __shared__ float lagAcc[2048];
    const int b  = blockIdx.y;
    const int t0 = blockIdx.x * 16;
    for (int i = threadIdx.x; i < L; i += 256) lagAcc[i] = 0.0f;
    __syncthreads();

    const int wave = threadIdx.x >> 5;
    const int lane = threadIdx.x & 31;
    const int r  = lane & 15;
    const int hi = lane >> 4;
    const _Float16* __restrict__ qrow = Q + ((size_t)b * strideQ + t0 + r) * 256;

    // hoist all Q fragments for this t-block (invariant over the s-loop)
    v16h afr[8];
#pragma unroll
    for (int c = 0; c < 8; ++c) afr[c] = af_load_a(qrow, c * 32, hi);

    const int nTiles = L >> 4;   // multiple of 2 (L = 2048 or 1536)
    for (int ts = wave * 2; ts < nTiles; ts += 16) {
        const int s0 = ts * 16;
        const _Float16* __restrict__ krow0 =
            Km + ((size_t)b * strideK + s0 + r) * 256 + hi * 16;
        const _Float16* __restrict__ krow1 = krow0 + 16 * 256;
        v8f acc0 = {}, acc1 = {};
#pragma unroll
        for (int c = 0; c < 8; ++c) {
            v16h b0 = *(const v16h*)(krow0 + c * 32);
            v16h b1 = *(const v16h*)(krow1 + c * 32);
            acc0 = AF_WMMA(afr[c], b0, acc0);
            acc1 = AF_WMMA(afr[c], b1, acc1);
        }
#pragma unroll
        for (int t = 0; t < 8; ++t) {
            const int tt = t0 + t + (hi ? 8 : 0);
            int lag0 = tt - (s0 + r);
            int lag1 = tt - (s0 + 16 + r);
            if (lag0 < 0) lag0 += L;
            if (lag1 < 0) lag1 += L;
            atomicAdd(&lagAcc[lag0], acc0[t]);
            atomicAdd(&lagAcc[lag1], acc1[t]);
        }
    }
    __syncthreads();
    for (int i = threadIdx.x; i < L; i += 256)
        atomicAdd(&meanv[(size_t)b * L + i], lagAcc[i]);
}

// ---------------------------------------------------------------------------
// top-k (k=7) + softmax per batch
// ---------------------------------------------------------------------------
__global__ __launch_bounds__(256) void k_topk(
    const float* __restrict__ meanv, float* __restrict__ wts, int* __restrict__ dly,
    int L, int ktop, float scale)
{
    __shared__ float vals[2048];
    __shared__ float rv[256];
    __shared__ int   ri[256];
    __shared__ float topv[8];
    __shared__ int   topi[8];
    const int b = blockIdx.x;
    for (int i = threadIdx.x; i < L; i += 256)
        vals[i] = meanv[(size_t)b * L + i] * scale;
    __syncthreads();
    for (int kk = 0; kk < ktop; ++kk) {
        float best = -3.4e38f; int bi = 0;
        for (int i = threadIdx.x; i < L; i += 256)
            if (vals[i] > best) { best = vals[i]; bi = i; }
        rv[threadIdx.x] = best; ri[threadIdx.x] = bi;
        __syncthreads();
        for (int s = 128; s > 0; s >>= 1) {
            if (threadIdx.x < s && rv[threadIdx.x + s] > rv[threadIdx.x]) {
                rv[threadIdx.x] = rv[threadIdx.x + s];
                ri[threadIdx.x] = ri[threadIdx.x + s];
            }
            __syncthreads();
        }
        if (threadIdx.x == 0) { topv[kk] = rv[0]; topi[kk] = ri[0]; vals[ri[0]] = -3.4e38f; }
        __syncthreads();
    }
    if (threadIdx.x == 0) {
        float m = topv[0];
        for (int i = 1; i < ktop; ++i) if (topv[i] > m) m = topv[i];
        float s = 0.0f;
        for (int i = 0; i < ktop; ++i) { float e = expf(topv[i] - m); topv[i] = e; s += e; }
        for (int i = 0; i < ktop; ++i) { wts[b * 8 + i] = topv[i] / s; dly[b * 8 + i] = topi[i]; }
    }
}

// out16[b,l,:] = sum_k w[b,k] * V[b,(l+delay_k)%L,:]
__global__ __launch_bounds__(256) void k_delay_agg(
    const _Float16* __restrict__ V, const float* __restrict__ wts,
    const int* __restrict__ dly, _Float16* __restrict__ out16,
    int L, int strideV, int ktop, int n)
{
    const size_t idx = (size_t)blockIdx.x * 256 + threadIdx.x;
    if (idx >= (size_t)n) return;
    const int d = (int)(idx & 255);
    const size_t bl = idx >> 8;
    const int l = (int)(bl % L);
    const int b = (int)(bl / L);
    float acc = 0.0f;
    for (int kk = 0; kk < ktop; ++kk) {
        int src = l + dly[b * 8 + kk];
        if (src >= L) src -= L;
        acc += wts[b * 8 + kk] * (float)V[((size_t)b * strideV + src) * 256 + d];
    }
    out16[((size_t)b * L + l) * 256 + d] = (_Float16)acc;
}

// ---------------------------------------------------------------------------
// series decomposition: seasonal = x - movmean25(x); optional trend write/acc
// accMode: 0=no trend, 1=trendAcc+=mm, 2=trendAcc=mm
// ---------------------------------------------------------------------------
__global__ __launch_bounds__(256) void k_decomp(
    const float* __restrict__ x, float* __restrict__ seasonal,
    float* __restrict__ trendAcc, int L, int D, int accMode, int n)
{
    const size_t idx = (size_t)blockIdx.x * 256 + threadIdx.x;
    if (idx >= (size_t)n) return;
    const int d = (int)(idx % D);
    const size_t ld = idx / D;
    const int l = (int)(ld % L);
    const int b = (int)(ld / L);
    float s = 0.0f;
    for (int j = -12; j <= 12; ++j) {
        int p = l + j;
        p = p < 0 ? 0 : (p >= L ? L - 1 : p);
        s += x[((size_t)b * L + p) * D + d];
    }
    const float mm = s * (1.0f / 25.0f);
    seasonal[idx] = x[idx] - mm;
    if (accMode == 1) trendAcc[idx] += mm;
    else if (accMode == 2) trendAcc[idx] = mm;
}

// mean over time: out[b,c] = mean_l x[b,l,c]
__global__ __launch_bounds__(256) void k_mean_time(
    const float* __restrict__ x, float* __restrict__ out, int L, int C)
{
    __shared__ float red[256];
    const int b = blockIdx.x / C;
    const int c = blockIdx.x % C;
    float s = 0.0f;
    for (int l = threadIdx.x; l < L; l += 256) s += x[((size_t)b * L + l) * C + c];
    red[threadIdx.x] = s; __syncthreads();
    for (int t = 128; t > 0; t >>= 1) {
        if (threadIdx.x < t) red[threadIdx.x] += red[threadIdx.x + t];
        __syncthreads();
    }
    if (threadIdx.x == 0) out[b * C + c] = red[0] / (float)L;
}

// seasonal_init / trend_init assembly, n = B*1536*7
__global__ __launch_bounds__(256) void k_build_inits(
    const float* __restrict__ seas7, const float* __restrict__ trend7,
    const float* __restrict__ meanx, float* __restrict__ sinit,
    float* __restrict__ trendb, int n)
{
    const size_t idx = (size_t)blockIdx.x * 256 + threadIdx.x;
    if (idx >= (size_t)n) return;
    const int c = (int)(idx % 7);
    const int l = (int)((idx / 7) % 1536);
    const int b = (int)(idx / (7 * 1536));
    if (l < 512) {
        sinit[idx]  = seas7[((size_t)b * 2048 + 1536 + l) * 7 + c];
        trendb[idx] = trend7[((size_t)b * 2048 + 1536 + l) * 7 + c];
    } else {
        sinit[idx]  = 0.0f;
        trendb[idx] = meanx[b * 7 + c];
    }
}

// embedding: circular conv3 over cin channels + mark @ time_w^T ; n=B*L*256
__global__ __launch_bounds__(256) void k_embed(
    const float* __restrict__ x, const float* __restrict__ tokenW,
    const float* __restrict__ mark, const float* __restrict__ timeW,
    float* __restrict__ out, int L, int cin, int n)
{
    const size_t idx = (size_t)blockIdx.x * 256 + threadIdx.x;
    if (idx >= (size_t)n) return;
    const int d = (int)(idx & 255);
    const size_t bl = idx >> 8;
    const int l = (int)(bl % L);
    const int b = (int)(bl / L);
    float acc = 0.0f;
    for (int j = 0; j < 3; ++j) {
        int p = l - 1 + j;
        if (p < 0) p += L;
        if (p >= L) p -= L;
        const float* __restrict__ xr = x + ((size_t)b * L + p) * cin;
        const float* __restrict__ wr = tokenW + ((size_t)d * cin) * 3 + j;
        for (int c = 0; c < cin; ++c) acc += xr[c] * wr[c * 3];
    }
    const float* __restrict__ mr = mark + ((size_t)b * L + l) * 4;
    for (int m = 0; m < 4; ++m) acc += mr[m] * timeW[d * 4 + m];
    out[idx] = acc;
}

// layernorm over feature dim (D=256), one block per (b,l)
__global__ __launch_bounds__(256) void k_ln_feat(
    const float* __restrict__ x, const float* __restrict__ g,
    const float* __restrict__ bb, float* __restrict__ out)
{
    __shared__ float red[256];
    const size_t bl = blockIdx.x;
    const int d = threadIdx.x;
    const float v = x[bl * 256 + d];
    red[d] = v; __syncthreads();
    for (int s = 128; s > 0; s >>= 1) { if (d < s) red[d] += red[d + s]; __syncthreads(); }
    const float mu = red[0] * (1.0f / 256.0f);
    __syncthreads();
    const float c = v - mu;
    red[d] = c * c; __syncthreads();
    for (int s = 128; s > 0; s >>= 1) { if (d < s) red[d] += red[d + s]; __syncthreads(); }
    const float var = red[0] * (1.0f / 256.0f);
    out[bl * 256 + d] = c * rsqrtf(var + 1e-5f) * g[d] + bb[d];
}

// subtract per-(b,d) mean over time; one block per (b,d)
__global__ __launch_bounds__(256) void k_sub_time_mean(
    const float* __restrict__ xin, float* __restrict__ xout, int L)
{
    __shared__ float red[256];
    const int b = blockIdx.x >> 8;
    const int d = blockIdx.x & 255;
    float s = 0.0f;
    for (int l = threadIdx.x; l < L; l += 256) s += xin[((size_t)b * L + l) * 256 + d];
    red[threadIdx.x] = s; __syncthreads();
    for (int t = 128; t > 0; t >>= 1) {
        if (threadIdx.x < t) red[threadIdx.x] += red[threadIdx.x + t];
        __syncthreads();
    }
    const float mu = red[0] / (float)L;
    for (int l = threadIdx.x; l < L; l += 256) {
        const size_t o = ((size_t)b * L + l) * 256 + d;
        xout[o] = xin[o] - mu;
    }
}

// trend conv: trendb[b,l,c] += circconv3(t123, tw[7,256,3]); n=B*1536*7
__global__ __launch_bounds__(256) void k_trend_conv(
    const float* __restrict__ t, const float* __restrict__ tw,
    float* __restrict__ trendb, int n)
{
    const size_t idx = (size_t)blockIdx.x * 256 + threadIdx.x;
    if (idx >= (size_t)n) return;
    const int c = (int)(idx % 7);
    const int l = (int)((idx / 7) % 1536);
    const int b = (int)(idx / (7 * 1536));
    float acc = 0.0f;
    for (int j = 0; j < 3; ++j) {
        int p = l - 1 + j;
        if (p < 0) p += 1536;
        if (p >= 1536) p -= 1536;
        const float* __restrict__ tr = t + ((size_t)b * 1536 + p) * 256;
        const float* __restrict__ wr = tw + ((size_t)c * 256) * 3 + j;
        for (int d = 0; d < 256; ++d) acc += tr[d] * wr[(size_t)d * 3];
    }
    trendb[idx] += acc;
}

// final: out[b,l,c] = trend[b,512+l,c] + proj(dec_norm[b,512+l,:]); n=B*1024*7
__global__ __launch_bounds__(256) void k_final(
    const float* __restrict__ decn, const float* __restrict__ trendb,
    const float* __restrict__ pw, const float* __restrict__ pb,
    float* __restrict__ out, int n)
{
    const size_t idx = (size_t)blockIdx.x * 256 + threadIdx.x;
    if (idx >= (size_t)n) return;
    const int c = (int)(idx % 7);
    const int l = (int)((idx / 7) % 1024);
    const int b = (int)(idx / (7 * 1024));
    const int l2 = 512 + l;
    const float* __restrict__ xr = decn + ((size_t)b * 1536 + l2) * 256;
    const float* __restrict__ wr = pw + (size_t)c * 256;
    float acc = pb[c];
    for (int d = 0; d < 256; ++d) acc += xr[d] * wr[d];
    out[idx] = trendb[((size_t)b * 1536 + l2) * 7 + c] + acc;
}

__global__ __launch_bounds__(256) void k_f32_to_f16(
    const float* __restrict__ src, _Float16* __restrict__ dst, int n)
{
    const size_t idx = (size_t)blockIdx.x * 256 + threadIdx.x;
    if (idx < (size_t)n) dst[idx] = (_Float16)src[idx];
}

__global__ __launch_bounds__(256) void k_fill(float* __restrict__ dst, float v, int n)
{
    const size_t idx = (size_t)blockIdx.x * 256 + threadIdx.x;
    if (idx < (size_t)n) dst[idx] = v;
}

// ===========================================================================
extern "C" void kernel_launch(void* const* d_in, const int* in_sizes, int n_in,
                              void* d_out, int out_size, void* d_ws, size_t ws_size,
                              hipStream_t stream)
{
    (void)in_sizes; (void)n_in; (void)out_size; (void)ws_size;
    const int B = 16, LS = 2048, LD = 1536, DM = 256, DFF = 1024, C = 7, KTOP = 7;
    const int ME = B * LS;   // 32768
    const int MD = B * LD;   // 24576

    auto F = [&](int i) { return (const float*)d_in[i]; };

    // ---- params (JAX pytree alphabetical flatten order) ----
    // 0..3: batch_x, batch_y, batch_x_mark, batch_y_mark
    const float* batch_x = F(0);
    const float* x_mark  = F(2);
    const float* y_mark  = F(3);
    // dec_emb: time_w=4, token_w=5
    // dec_layers[0]: cross k.b6 k.w7 o.b8 o.w9 q.b10 q.w11 v.b12 v.w13
    //                self  k.b14 k.w15 o.b16 o.w17 q.b18 q.w19 v.b20 v.w21
    //                trend_w=22 w1=23 w2=24
    // dec_norm: b=25 g=26 ; enc_emb: time_w=27 token_w=28
    // enc_layers[0]: k.b29 k.w30 o.b31 o.w32 q.b33 q.w34 v.b35 v.w36 w1=37 w2=38
    // enc_layers[1]: 39..48 ; enc_norm: b=49 g=50 ; proj: b=51 w=52

    // ---- workspace carve ----
    char* ws = (char*)d_ws;
    size_t off = 0;
    auto carve = [&](size_t bytes) {
        size_t o = off;
        off += (bytes + 255) & ~(size_t)255;
        return (void*)(ws + o);
    };
    float*    ENC_X   = (float*)carve((size_t)ME * DM * 4);
    float*    DEC_X   = (float*)carve((size_t)MD * DM * 4);
    float*    TMPF    = (float*)carve((size_t)ME * DM * 4);
    float*    T123    = (float*)carve((size_t)MD * DM * 4);
    _Float16* XF16    = (_Float16*)carve((size_t)ME * DM * 2);
    char*     UNION_  = (char*)carve((size_t)ME * DFF * 2);        // FF16 / Q16K16V16
    _Float16* FF16    = (_Float16*)UNION_;
    _Float16* Q16     = (_Float16*)UNION_;
    _Float16* K16     = (_Float16*)(UNION_ + (size_t)ME * DM * 2);
    _Float16* V16     = (_Float16*)(UNION_ + (size_t)2 * ME * DM * 2);
    _Float16* ENCF16  = (_Float16*)carve((size_t)ME * DM * 2);
    float*    SEAS7   = (float*)carve((size_t)B * LS * C * 4);
    float*    TREND7  = (float*)carve((size_t)B * LS * C * 4);
    float*    MEANX   = (float*)carve((size_t)B * C * 4);
    float*    SINIT   = (float*)carve((size_t)B * LD * C * 4);
    float*    TRENDB  = (float*)carve((size_t)B * LD * C * 4);
    float*    MEANV   = (float*)carve((size_t)B * LS * 4);
    float*    WTS     = (float*)carve((size_t)B * 8 * 4);
    int*      DLY     = (int*)carve((size_t)B * 8 * 4);
    _Float16* WH      = (_Float16*)carve((size_t)3000000 * 2);

    // f16 weight pool layout
    size_t wp = 0;
    auto wslot = [&](size_t nel) { _Float16* r = WH + wp; wp += nel; return r; };
    _Float16 *e0qw = wslot(65536), *e0kw = wslot(65536), *e0vw = wslot(65536), *e0ow = wslot(65536);
    _Float16 *e0w1 = wslot(262144), *e0w2 = wslot(262144);
    _Float16 *e1qw = wslot(65536), *e1kw = wslot(65536), *e1vw = wslot(65536), *e1ow = wslot(65536);
    _Float16 *e1w1 = wslot(262144), *e1w2 = wslot(262144);
    _Float16 *dsqw = wslot(65536), *dskw = wslot(65536), *dsvw = wslot(65536), *dsow = wslot(65536);
    _Float16 *dcqw = wslot(65536), *dckw = wslot(65536), *dcvw = wslot(65536), *dcow = wslot(65536);
    _Float16 *dw1 = wslot(262144), *dw2 = wslot(262144);

    auto gblk = [](int n) { return (unsigned)((n + 255) / 256); };
    auto cvt = [&](const float* s, _Float16* d, int n) {
        k_f32_to_f16<<<gblk(n), 256, 0, stream>>>(s, d, n);
    };
    auto fill = [&](float* d, float v, int n) {
        k_fill<<<gblk(n), 256, 0, stream>>>(d, v, n);
    };
    auto gemm = [&](const _Float16* A, int M, int K, const _Float16* W,
                    const float* bias, int N, const float* resid,
                    float* outF, _Float16* outH, int gelu) {
        dim3 g((unsigned)((M / 64) * (N / 64)));
        if (K == 1024)
            k_gemm_wmma<1024><<<g, 256, 0, stream>>>(A, W, bias, resid, outF, outH, M, N, gelu);
        else
            k_gemm_wmma<256><<<g, 256, 0, stream>>>(A, W, bias, resid, outF, outH, M, N, gelu);
    };

    // ---- convert GEMM weights to f16 ----
    cvt(F(34), e0qw, 65536); cvt(F(30), e0kw, 65536); cvt(F(36), e0vw, 65536); cvt(F(32), e0ow, 65536);
    cvt(F(37), e0w1, 262144); cvt(F(38), e0w2, 262144);
    cvt(F(44), e1qw, 65536); cvt(F(40), e1kw, 65536); cvt(F(46), e1vw, 65536); cvt(F(42), e1ow, 65536);
    cvt(F(47), e1w1, 262144); cvt(F(48), e1w2, 262144);
    cvt(F(19), dsqw, 65536); cvt(F(15), dskw, 65536); cvt(F(21), dsvw, 65536); cvt(F(17), dsow, 65536);
    cvt(F(11), dcqw, 65536); cvt(F(7),  dckw, 65536); cvt(F(13), dcvw, 65536); cvt(F(9),  dcow, 65536);
    cvt(F(23), dw1, 262144); cvt(F(24), dw2, 262144);

    // ---- encoder-side init ----
    k_decomp<<<gblk(B * LS * C), 256, 0, stream>>>(batch_x, SEAS7, TREND7, LS, C, 2, B * LS * C);
    k_mean_time<<<B * C, 256, 0, stream>>>(batch_x, MEANX, LS, C);
    k_build_inits<<<gblk(B * LD * C), 256, 0, stream>>>(SEAS7, TREND7, MEANX, SINIT, TRENDB, B * LD * C);
    k_embed<<<gblk(ME * DM), 256, 0, stream>>>(batch_x, F(28), x_mark, F(27), ENC_X, LS, C, ME * DM);

    struct AttnW {
        const float* qb; const _Float16* qw; const float* kb; const _Float16* kw;
        const float* vb; const _Float16* vw; const float* ob; const _Float16* ow;
    };
    AttnW encA[2] = {
        { F(33), e0qw, F(29), e0kw, F(35), e0vw, F(31), e0ow },
        { F(43), e1qw, F(39), e1kw, F(45), e1vw, F(41), e1ow } };
    const _Float16* encW1[2] = { e0w1, e1w1 };
    const _Float16* encW2[2] = { e0w2, e1w2 };

    // self-attention block: X (f32, [B,L,256]) updated in place via TMPF
    auto self_attn = [&](float* X, int L, int M, const AttnW& w,
                         float* trendAcc, int accMode) {
        cvt(X, XF16, M * DM);
        gemm(XF16, M, DM, w.qw, w.qb, DM, nullptr, nullptr, Q16, 0);
        gemm(XF16, M, DM, w.kw, w.kb, DM, nullptr, nullptr, K16, 0);
        gemm(XF16, M, DM, w.vw, w.vb, DM, nullptr, nullptr, V16, 0);
        fill(MEANV, 0.0f, B * L);
        k_corr_wmma<<<dim3((unsigned)(L / 16), B), 256, 0, stream>>>(Q16, K16, MEANV, L, L, L);
        k_topk<<<B, 256, 0, stream>>>(MEANV, WTS, DLY, L, KTOP, 1.0f / 256.0f);
        k_delay_agg<<<gblk(M * DM), 256, 0, stream>>>(V16, WTS, DLY, XF16, L, L, KTOP, M * DM);
        gemm(XF16, M, DM, w.ow, w.ob, DM, X, TMPF, nullptr, 0);
        k_decomp<<<gblk(M * DM), 256, 0, stream>>>(TMPF, X, trendAcc, L, DM, accMode, M * DM);
    };
    auto ffn = [&](float* X, int L, int M, const _Float16* w1h, const _Float16* w2h,
                   float* trendAcc, int accMode) {
        cvt(X, XF16, M * DM);
        gemm(XF16, M, DM, w1h, nullptr, DFF, nullptr, nullptr, FF16, 1);
        gemm(FF16, M, DFF, w2h, nullptr, DM, X, TMPF, nullptr, 0);
        k_decomp<<<gblk(M * DM), 256, 0, stream>>>(TMPF, X, trendAcc, L, DM, accMode, M * DM);
    };

    // ---- encoder layers ----
    for (int li = 0; li < 2; ++li) {
        self_attn(ENC_X, LS, ME, encA[li], nullptr, 0);
        ffn(ENC_X, LS, ME, encW1[li], encW2[li], nullptr, 0);
    }
    // enc norm (my_layernorm)
    k_ln_feat<<<ME, 256, 0, stream>>>(ENC_X, F(50), F(49), TMPF);
    k_sub_time_mean<<<B * DM, 256, 0, stream>>>(TMPF, ENC_X, LS);
    cvt(ENC_X, ENCF16, ME * DM);

    // ---- decoder ----
    k_embed<<<gblk(MD * DM), 256, 0, stream>>>(SINIT, F(5), y_mark, F(4), DEC_X, LD, C, MD * DM);
    fill(T123, 0.0f, MD * DM);

    AttnW decSelf  = { F(18), dsqw, F(14), dskw, F(20), dsvw, F(16), dsow };
    self_attn(DEC_X, LD, MD, decSelf, T123, 1);

    // cross attention: Q from dec, K/V from enc_out (truncated to first LD rows)
    cvt(DEC_X, XF16, MD * DM);
    gemm(XF16,   MD, DM, dcqw, F(10), DM, nullptr, nullptr, Q16, 0);
    gemm(ENCF16, ME, DM, dckw, F(6),  DM, nullptr, nullptr, K16, 0);
    gemm(ENCF16, ME, DM, dcvw, F(12), DM, nullptr, nullptr, V16, 0);
    fill(MEANV, 0.0f, B * LD);
    k_corr_wmma<<<dim3((unsigned)(LD / 16), B), 256, 0, stream>>>(Q16, K16, MEANV, LD, LD, LS);
    k_topk<<<B, 256, 0, stream>>>(MEANV, WTS, DLY, LD, KTOP, 1.0f / 256.0f);
    k_delay_agg<<<gblk(MD * DM), 256, 0, stream>>>(V16, WTS, DLY, XF16, LD, LS, KTOP, MD * DM);
    gemm(XF16, MD, DM, dcow, F(8), DM, DEC_X, TMPF, nullptr, 0);
    k_decomp<<<gblk(MD * DM), 256, 0, stream>>>(TMPF, DEC_X, T123, LD, DM, 1, MD * DM);

    ffn(DEC_X, LD, MD, dw1, dw2, T123, 1);

    // trend = trend_init + conv_circ(t1+t2+t3, trend_w)
    k_trend_conv<<<gblk(B * LD * C), 256, 0, stream>>>(T123, F(22), TRENDB, B * LD * C);

    // dec norm + final projection + slice
    k_ln_feat<<<MD, 256, 0, stream>>>(DEC_X, F(26), F(25), TMPF);
    k_sub_time_mean<<<B * DM, 256, 0, stream>>>(TMPF, DEC_X, LD);
    k_final<<<gblk(B * 1024 * C), 256, 0, stream>>>(DEC_X, TRENDB, F(52), F(51),
                                                    (float*)d_out, B * 1024 * C);
}
